// ScaledDotProductAttention_25194278158767
// MI455X (gfx1250) — matrix operand order repaired and checked by reading, not compiled, not run on hardware
//
#include <hip/hip_runtime.h>

// Reference problem: B=2, H=16, S=2048, d=64, causal+pad mask, NEG=1e9 (finite).
#define BD   2
#define HH   16
#define SS   2048
#define DKD  64
#define BM   64      // query rows per block (4 waves x 16)
#define BN   64      // key rows per iteration (4 WMMA N-tiles, 2 K-chunks for PV)
#define NW   4       // waves per block
#define LDP  72      // LDS row stride (bf16): 144B = 36 dwords -> 16 distinct bank groups

typedef __attribute__((ext_vector_type(2)))  __bf16 v2bf;
typedef __attribute__((ext_vector_type(8)))  __bf16 v8bf;
typedef __attribute__((ext_vector_type(16))) __bf16 v16bf;
typedef __attribute__((ext_vector_type(8)))  float  v8f;
typedef __attribute__((ext_vector_type(4)))  float  fvec4;

// 16-element bf16 fragment from two 16-byte LDS chunks (2x ds_load_b128, conflict-free).
__device__ __forceinline__ v16bf frag2(const __bf16* p0, const __bf16* p1) {
  v8bf lo = *(const v8bf*)p0;
  v8bf hi = *(const v8bf*)p1;
  v16bf r;
#pragma unroll
  for (int i = 0; i < 8; ++i) { r[i] = lo[i]; r[i + 8] = hi[i]; }
  return r;
}

__device__ __forceinline__ v2bf cvt2(float a, float b) {  // packed f32->bf16 pair
  v2bf r; r[0] = (__bf16)a; r[1] = (__bf16)b; return r;
}

// Reductions across a 16-lane half-wave (xor masks < 16 stay inside the half).
__device__ __forceinline__ float redmax16(float x) {
#pragma unroll
  for (int m = 8; m; m >>= 1) x = fmaxf(x, __shfl_xor(x, m, 32));
  return x;
}
__device__ __forceinline__ float redsum16(float x) {
#pragma unroll
  for (int m = 8; m; m >>= 1) x += __shfl_xor(x, m, 32);
  return x;
}

__global__ __launch_bounds__(128)
void fa_fwd_cdna5(const float* __restrict__ Q, const float* __restrict__ K,
                  const float* __restrict__ V, const int* __restrict__ padm,
                  float* __restrict__ O) {
  __shared__ __align__(16) __bf16 Qs[BM][LDP];       // 64 q-rows x 64 d (row-major)
  __shared__ __align__(16) __bf16 Ks[BN][LDP];       // 64 k-rows x 64 d (row-major -> K^T frags)
  __shared__ __align__(16) __bf16 Vt[DKD][LDP];      // transposed: 64 d x 64 keys (-> V B-frags)
  __shared__ __align__(16) __bf16 Ps[NW][16][LDP];   // per-wave P scratch, 16 x 64 keys
  __shared__ int pads[BN];

  const int tid  = threadIdx.x;
  const int wave = tid >> 5;
  const int lane = tid & 31;
  const int half = lane >> 4;     // C/D layout: lanes 0-15 -> rows r, 16-31 -> rows r+8
  const int ncol = lane & 15;     // N column within a 16-wide tile
  const int bh   = blockIdx.y;
  const int b    = bh / HH;
  const int qbase = blockIdx.x * BM;
  const long base = (long)bh * SS * DKD;
  const float* Qg = Q + base;
  const float* Kg = K + base;
  const float* Vg = V + base;

  // ---- Stage Q block (64x64 f32 -> bf16 LDS), 8 float4 per thread ----
#pragma unroll
  for (int i = 0; i < 8; ++i) {
    const int e = (tid + i * 128) * 4;
    const int r = e >> 6, c = e & 63;
    fvec4 q4 = *(const fvec4*)(Qg + (long)(qbase + r) * DKD + c);
    *(v2bf*)&Qs[r][c]     = cvt2(q4[0], q4[1]);
    *(v2bf*)&Qs[r][c + 2] = cvt2(q4[2], q4[3]);
  }
  __syncthreads();

  // ---- Per-wave Q A-fragments, kept in VGPRs across all key blocks ----
  // A layout (16x32): lanes 0-15 row M=lane hold K=0..7,16..23; lanes 16-31 hold K=8..15,24..31.
  const int arow = wave * 16 + ncol;
  const int asel = half * 8;
  const v16bf qf0 = frag2(&Qs[arow][asel],      &Qs[arow][asel + 16]);       // d 0..31
  const v16bf qf1 = frag2(&Qs[arow][32 + asel], &Qs[arow][32 + asel + 16]);  // d 32..63

  float m_r[8], l_r[8];
#pragma unroll
  for (int r = 0; r < 8; ++r) { m_r[r] = -1e30f; l_r[r] = 0.0f; }
  v8f acc[4] = {};   // 4 d-tiles of 16 -> full d=64 output accumulator

  const int kb  = half * 8;                 // B-frag K-range base (lanes 16-31 hold K=8..15,24..31)
  const int nkb = qbase / BN + 1;           // causal early exit: jbase <= qbase+63

  for (int jb = 0; jb < nkb; ++jb) {
    const int jbase = jb * BN;
    __syncthreads();   // previous iteration finished reading Ks/Vt

    // ---- Stage K (row-major, packed cvt) and V (transposed): 8 float4 per thread each ----
#pragma unroll
    for (int i = 0; i < 8; ++i) {
      const int e = (tid + i * 128) * 4;
      const int r = e >> 6, c = e & 63;
      fvec4 k4 = *(const fvec4*)(Kg + (long)(jbase + r) * DKD + c);
      fvec4 v4 = *(const fvec4*)(Vg + (long)(jbase + r) * DKD + c);
      *(v2bf*)&Ks[r][c]     = cvt2(k4[0], k4[1]);
      *(v2bf*)&Ks[r][c + 2] = cvt2(k4[2], k4[3]);
#pragma unroll
      for (int j = 0; j < 4; ++j) Vt[c + j][r] = (__bf16)v4[j];
    }
    if (tid < BN) pads[tid] = padm[b * SS + jbase + tid];
    __syncthreads();

    // ---- Scores S = Q * K^T : four 16-key tiles, accumulate over d in 2 chunks ----
    v8f s[4];
#pragma unroll
    for (int t = 0; t < 4; ++t) {
      v8f st = {};
      v16bf kf = frag2(&Ks[t * 16 + ncol][kb], &Ks[t * 16 + ncol][kb + 16]);
      st = __builtin_amdgcn_wmma_f32_16x16x32_bf16(false, qf0, false, kf, (short)0, st, false, false);
      kf = frag2(&Ks[t * 16 + ncol][32 + kb], &Ks[t * 16 + ncol][32 + kb + 16]);
      st = __builtin_amdgcn_wmma_f32_16x16x32_bf16(false, qf1, false, kf, (short)0, st, false, false);
      s[t] = st;
    }

    // ---- Mask + online softmax (row r lives in a 16-lane half-wave) ----
    bool pf[4]; int kg[4];
#pragma unroll
    for (int t = 0; t < 4; ++t) {
      pf[t] = pads[t * 16 + ncol] != 0;
      kg[t] = jbase + t * 16 + ncol;
    }
#pragma unroll
    for (int r = 0; r < 8; ++r) {
      const int qg = qbase + wave * 16 + half * 8 + r;
      float x[4];
#pragma unroll
      for (int t = 0; t < 4; ++t)
        x[t] = s[t][r] * 0.125f - (((kg[t] > qg) || pf[t]) ? 1.0e9f : 0.0f);
      const float rm   = redmax16(fmaxf(fmaxf(x[0], x[1]), fmaxf(x[2], x[3])));
      const float mnew = fmaxf(m_r[r], rm);
      float e[4], esum = 0.0f;
#pragma unroll
      for (int t = 0; t < 4; ++t) { e[t] = __expf(x[t] - mnew); esum += e[t]; }
      const float corr = __expf(m_r[r] - mnew);
      l_r[r] = l_r[r] * corr + redsum16(esum);
      m_r[r] = mnew;
#pragma unroll
      for (int t = 0; t < 4; ++t) acc[t][r] *= corr;
#pragma unroll
      for (int t = 0; t < 4; ++t)
        Ps[wave][half * 8 + r][t * 16 + ncol] = (__bf16)e[t];
    }
    // Per-wave LDS scratch: order the P stores before the A-fragment reloads.
    asm volatile("s_wait_dscnt 0x0" ::: "memory");

    // ---- O += P * V : A = P (16x64 keys, 2 chunks), B = Vt rows (keys x 16 d-cols) ----
    const v16bf pf0 = frag2(&Ps[wave][ncol][asel],      &Ps[wave][ncol][asel + 16]);       // keys 0..31
    const v16bf pf1 = frag2(&Ps[wave][ncol][32 + asel], &Ps[wave][ncol][32 + asel + 16]);  // keys 32..63
#pragma unroll
    for (int t = 0; t < 4; ++t) {
      v16bf vf = frag2(&Vt[t * 16 + ncol][kb], &Vt[t * 16 + ncol][kb + 16]);
      acc[t] = __builtin_amdgcn_wmma_f32_16x16x32_bf16(false, pf0, false, vf, (short)0, acc[t], false, false);
      vf = frag2(&Vt[t * 16 + ncol][32 + kb], &Vt[t * 16 + ncol][32 + kb + 16]);
      acc[t] = __builtin_amdgcn_wmma_f32_16x16x32_bf16(false, pf1, false, vf, (short)0, acc[t], false, false);
    }
  }

  // ---- Normalize and store (f32 output, [B,H,S,64] row-major) ----
  float* Og = O + base;
#pragma unroll
  for (int r = 0; r < 8; ++r) {
    const float inv = 1.0f / l_r[r];
    const long row = qbase + wave * 16 + half * 8 + r;
#pragma unroll
    for (int t = 0; t < 4; ++t)
      Og[row * DKD + t * 16 + ncol] = acc[t][r] * inv;
  }
}

extern "C" void kernel_launch(void* const* d_in, const int* in_sizes, int n_in,
                              void* d_out, int out_size, void* d_ws, size_t ws_size,
                              hipStream_t stream) {
  (void)in_sizes; (void)n_in; (void)out_size; (void)d_ws; (void)ws_size;
  const float* Q   = (const float*)d_in[0];
  const float* K   = (const float*)d_in[1];
  const float* V   = (const float*)d_in[2];
  const int*   pad = (const int*)d_in[3];
  float*       out = (float*)d_out;
  dim3 grid(SS / BM, BD * HH);   // (32, 32)
  fa_fwd_cdna5<<<grid, 128, 0, stream>>>(Q, K, V, pad, out);
}
